// EqLayer_88656714925230
// MI455X (gfx1250) — matrix operand order repaired and checked by reading, not compiled, not run on hardware
//
#include <hip/hip_runtime.h>

typedef __attribute__((ext_vector_type(16))) _Float16 v16h;
typedef __attribute__((ext_vector_type(8)))  float    v8f;

union Frag  { unsigned u[8]; v16h h; };
union Pack2 { _Float16 h[2]; unsigned u; };

#define N_SCALARS 32
#define HIDDEN    128
#define X_DIM     96            // 32 + 8*4*2
#define MSG_IN    256           // 2*96 + 64
#define TILE_E    128
#define ROWA      132           // padded dwords per 256-f16 row (4-bank shift/row)
#define ROWW2     68            // padded dwords per 128-f16 row

__device__ __forceinline__ v8f wmma_f16(const Frag& a, const Frag& b, v8f c) {
    return __builtin_amdgcn_wmma_f32_16x16x32_f16(
        false, a.h, false, b.h, (short)0, c, false, false);
}

__global__ void zero_f32_kernel(float* p, int n) {
    int i = blockIdx.x * 256 + threadIdx.x;
    if (i < n) p[i] = 0.0f;
}

// W1 (256x128 k-major) -> W1t f16 [128][256] (n-major, k contiguous)
// W2 (128x96  k-major) -> W2t f16 [96][128]
__global__ void convert_w_kernel(const float* W1, const float* W2,
                                 _Float16* W1t, _Float16* W2t) {
    int i = blockIdx.x * 256 + threadIdx.x;
    if (i < MSG_IN * HIDDEN) {
        int n = i >> 8, k = i & 255;
        W1t[n * MSG_IN + k] = (_Float16)W1[k * HIDDEN + n];
    }
    if (i < HIDDEN * X_DIM) {
        int n = i >> 7, k = i & 127;
        W2t[n * HIDDEN + k] = (_Float16)W2[k * X_DIM + n];
    }
}

__global__ __launch_bounds__(256) void eq_main_kernel(
    const float* __restrict__ xs,      // (N,32)
    const float* __restrict__ xr,      // (N,8,8)
    const int*   __restrict__ ei,      // (2,E): row = ei[0..E), col = ei[E..2E)
    const float* __restrict__ dist,    // (E,64)
    const float* __restrict__ rot,     // (E,4,2,2)
    const _Float16* __restrict__ W1t,  // [128][256] f16
    const float* __restrict__ b1,      // (128)
    const _Float16* __restrict__ W2t,  // [96][128] f16
    const float* __restrict__ b2,      // (96)
    float* __restrict__ out,           // out_scalar (N*32) ++ out_rot (N*64)
    int E, int N)
{
    __shared__ unsigned sAd[TILE_E * ROWA];   // 66 KB message tile (reused for h)
    __shared__ unsigned sW1[HIDDEN * ROWA];   // 66 KB W1t staged
    __shared__ unsigned sW2[X_DIM * ROWW2];   // 25.5 KB W2t staged

    const int tid      = threadIdx.x;
    const int tileBase = blockIdx.x * TILE_E;
    _Float16* sAh = (_Float16*)sAd;

    // ---------- stage weights into LDS (vectorized b128 copies) ----------
    {
        const uint4* W1v = (const uint4*)W1t;      // 4096 uint4
        #pragma unroll 4
        for (int it = 0; it < 16; ++it) {
            int idx = it * 256 + tid;
            int n = idx >> 5, d4 = (idx & 31) << 2;
            *(uint4*)(sW1 + n * ROWA + d4) = W1v[idx];
        }
        const uint4* W2v = (const uint4*)W2t;      // 1536 uint4
        #pragma unroll
        for (int it = 0; it < 6; ++it) {
            int idx = it * 256 + tid;
            int n = idx >> 4, d4 = (idx & 15) << 2;
            *(uint4*)(sW2 + n * ROWW2 + d4) = W2v[idx];
        }
    }

    // ---------- Phase 0a: scalar + distance features (float4 loads) ----------
    #pragma unroll 4
    for (int it = 0; it < 16; ++it) {
        int idx = it * 256 + tid;                 // 128 edges * 32 groups-of-4
        int e = idx >> 5, g = idx & 31;
        int eg = tileBase + e;
        float4 v = {0.f, 0.f, 0.f, 0.f};
        int dcol0;
        if (g < 8) {
            dcol0 = g << 2;
            if (eg < E) v = *(const float4*)(xs + (size_t)ei[E + eg] * 32 + dcol0);
        } else if (g < 16) {
            int c0 = (g - 8) << 2; dcol0 = 96 + c0;
            if (eg < E) v = *(const float4*)(xs + (size_t)ei[eg] * 32 + c0);
        } else {
            int c0 = (g - 16) << 2; dcol0 = 192 + c0;
            if (eg < E) v = *(const float4*)(dist + (size_t)eg * 64 + c0);
        }
        Pack2 p0, p1;
        p0.h[0] = (_Float16)v.x; p0.h[1] = (_Float16)v.y;
        p1.h[0] = (_Float16)v.z; p1.h[1] = (_Float16)v.w;
        uint2 pk; pk.x = p0.u; pk.y = p1.u;
        *(uint2*)(sAd + e * ROWA + (dcol0 >> 1)) = pk;
    }

    // ---------- Phase 0b: inverse-rotated features ----------
    // out[l] = xr0*rot[k][l][0] + xr1*rot[k][l][1]  (R^T applied)
    #pragma unroll 2
    for (int it = 0; it < 8; ++it) {
        int idx = it * 256 + tid;                 // 128 edges * 2 parts * 8 reps
        int e = idx >> 4, r = idx & 15;
        int part = r >> 3, j = r & 7;
        int eg = tileBase + e;
        uint4 pk = {0u, 0u, 0u, 0u};
        if (eg < E) {
            int node = part ? ei[eg] : ei[E + eg];   // part0=dst(col), part1=src(row)
            const float4* xr4 = (const float4*)(xr + (size_t)node * 64 + j * 8);
            float4 A0 = xr4[0], A1 = xr4[1];
            const float4* r4 = (const float4*)(rot + (size_t)eg * 16);
            float4 R0 = r4[0], R1 = r4[1], R2 = r4[2], R3 = r4[3];
            Pack2 q0, q1, q2, q3;
            q0.h[0] = (_Float16)(A0.x * R0.x + A0.y * R0.y);
            q0.h[1] = (_Float16)(A0.x * R0.z + A0.y * R0.w);
            q1.h[0] = (_Float16)(A0.z * R1.x + A0.w * R1.y);
            q1.h[1] = (_Float16)(A0.z * R1.z + A0.w * R1.w);
            q2.h[0] = (_Float16)(A1.x * R2.x + A1.y * R2.y);
            q2.h[1] = (_Float16)(A1.x * R2.z + A1.y * R2.w);
            q3.h[0] = (_Float16)(A1.z * R3.x + A1.w * R3.y);
            q3.h[1] = (_Float16)(A1.z * R3.z + A1.w * R3.w);
            pk.x = q0.u; pk.y = q1.u; pk.z = q2.u; pk.w = q3.u;
        }
        int d0 = (part ? 64 : 16) + j * 4;        // dword offset of cols base..base+7
        *(uint4*)(sAd + e * ROWA + d0) = pk;
    }
    __syncthreads();

    // Wave decomposition: 2 M-tiles x 4 N-tiles per wave (B frags reused 2x)
    const int w   = tid >> 5;
    const int l   = tid & 31;
    const int m16 = l & 15;
    const int hi  = (l >> 4) & 1;
    const int mp  = w & 3;           // M-pair: M-tiles 2mp, 2mp+1 (rows mp*32..+31)
    const int nh  = w >> 2;          // N-half

    const unsigned* rowA0 = sAd + (mp * 32 + m16) * ROWA;
    const unsigned* rowA1 = rowA0 + 16 * ROWA;

    // ---------- GEMM1: (32x256) @ W1t[nh half] -> silu -> h (32x64) ----------
    v8f acc[2][4];
    #pragma unroll
    for (int mt = 0; mt < 2; ++mt)
        #pragma unroll
        for (int nt = 0; nt < 4; ++nt) acc[mt][nt] = (v8f){0,0,0,0,0,0,0,0};

    {
        Frag a0, a1, an0, an1, b, bn;
        #pragma unroll
        for (int v = 0; v < 8; ++v) {
            int d = (v >> 2) * 8 + hi * 4 + (v & 3);
            a0.u[v] = rowA0[d];
            a1.u[v] = rowA1[d];
        }
        #pragma unroll
        for (int ks = 0; ks < 8; ++ks) {
            if (ks < 7) {
                #pragma unroll
                for (int v = 0; v < 8; ++v) {
                    int d = (ks + 1) * 16 + (v >> 2) * 8 + hi * 4 + (v & 3);
                    an0.u[v] = rowA0[d];
                    an1.u[v] = rowA1[d];
                }
            }
            #pragma unroll
            for (int v = 0; v < 8; ++v)
                b.u[v] = sW1[((nh * 4) * 16 + m16) * ROWA + ks * 16 + (v >> 2) * 8 + hi * 4 + (v & 3)];
            #pragma unroll
            for (int nt = 0; nt < 4; ++nt) {
                if (nt < 3) {
                    #pragma unroll
                    for (int v = 0; v < 8; ++v)
                        bn.u[v] = sW1[((nh * 4 + nt + 1) * 16 + m16) * ROWA + ks * 16 + (v >> 2) * 8 + hi * 4 + (v & 3)];
                }
                acc[0][nt] = wmma_f16(a0, b, acc[0][nt]);
                acc[1][nt] = wmma_f16(a1, b, acc[1][nt]);
                b = bn;
            }
            a0 = an0; a1 = an1;
        }
    }

    // bias + silu, restage h (rows mp*32..+31, cols nh*64..+63) into LDS
    #pragma unroll
    for (int nt = 0; nt < 4; ++nt) {
        int n = (nh * 4 + nt) * 16 + m16;
        float bias = b1[n];
        #pragma unroll
        for (int mt = 0; mt < 2; ++mt) {
            #pragma unroll
            for (int v = 0; v < 8; ++v) {
                int m = v + 8 * hi;
                float x = acc[mt][nt][v] + bias;
                float s = x / (1.0f + __expf(-x));    // silu
                sAh[(mp * 32 + mt * 16 + m) * (ROWA * 2) + n] = (_Float16)s;
            }
        }
    }
    __syncthreads();

    // ---------- GEMM2: h (32x128) @ W2t[nh third-pair] -> x_out (32x48) ----------
    v8f acc2[2][3];
    #pragma unroll
    for (int mt = 0; mt < 2; ++mt)
        #pragma unroll
        for (int nt = 0; nt < 3; ++nt) acc2[mt][nt] = (v8f){0,0,0,0,0,0,0,0};

    {
        Frag a0, a1, an0, an1, b, bn;
        #pragma unroll
        for (int v = 0; v < 8; ++v) {
            int d = (v >> 2) * 8 + hi * 4 + (v & 3);
            a0.u[v] = rowA0[d];
            a1.u[v] = rowA1[d];
        }
        #pragma unroll
        for (int ks = 0; ks < 4; ++ks) {
            if (ks < 3) {
                #pragma unroll
                for (int v = 0; v < 8; ++v) {
                    int d = (ks + 1) * 16 + (v >> 2) * 8 + hi * 4 + (v & 3);
                    an0.u[v] = rowA0[d];
                    an1.u[v] = rowA1[d];
                }
            }
            #pragma unroll
            for (int v = 0; v < 8; ++v)
                b.u[v] = sW2[((nh * 3) * 16 + m16) * ROWW2 + ks * 16 + (v >> 2) * 8 + hi * 4 + (v & 3)];
            #pragma unroll
            for (int nt = 0; nt < 3; ++nt) {
                if (nt < 2) {
                    #pragma unroll
                    for (int v = 0; v < 8; ++v)
                        bn.u[v] = sW2[((nh * 3 + nt + 1) * 16 + m16) * ROWW2 + ks * 16 + (v >> 2) * 8 + hi * 4 + (v & 3)];
                }
                acc2[0][nt] = wmma_f16(a0, b, acc2[0][nt]);
                acc2[1][nt] = wmma_f16(a1, b, acc2[1][nt]);
                b = bn;
            }
            a0 = an0; a1 = an1;
        }
    }

    // ---------- Epilogue: bias, forward rotation, scatter-add ----------
    float* out_scalar = out;
    float* out_rot    = out + (size_t)N * 32;

    int colm[2][8];
    bool okm[2][8];
    #pragma unroll
    for (int mt = 0; mt < 2; ++mt) {
        #pragma unroll
        for (int v = 0; v < 8; ++v) {
            int eg = tileBase + mp * 32 + mt * 16 + v + 8 * hi;
            okm[mt][v]  = (eg < E);
            colm[mt][v] = ei[E + (okm[mt][v] ? eg : (E - 1))];
        }
    }

    #pragma unroll
    for (int nt = 0; nt < 3; ++nt) {
        int n = (nh * 3 + nt) * 16 + m16;
        float bias = b2[n];
        #pragma unroll
        for (int mt = 0; mt < 2; ++mt) {
            #pragma unroll
            for (int v = 0; v < 8; ++v) {
                int eg  = tileBase + mp * 32 + mt * 16 + v + 8 * hi;
                int egc = okm[mt][v] ? eg : (E - 1);
                float val = acc2[mt][nt][v] + bias;
                if (n < 32) {                       // uniform: nh==0 && nt<2
                    if (okm[mt][v])
                        atomicAdd(out_scalar + (size_t)colm[mt][v] * 32 + n, val);
                } else {
                    // (m=0,m=1) pair of each 2-vector sits in adjacent lanes
                    int q = n - 32;
                    int j = q >> 3, kk = (q >> 1) & 3, p = q & 1;   // p == lane parity
                    float partner = __shfl_xor(val, 1, 32);
                    float mr0 = p ? partner : val;
                    float mr1 = p ? val : partner;
                    const float* rf = rot + (size_t)egc * 16 + kk * 4;
                    float o = mr0 * rf[p] + mr1 * rf[2 + p];
                    if (okm[mt][v])
                        atomicAdd(out_rot + (size_t)colm[mt][v] * 64 + j * 8 + kk * 2 + p, o);
                }
            }
        }
    }
}

extern "C" void kernel_launch(void* const* d_in, const int* in_sizes, int n_in,
                              void* d_out, int out_size, void* d_ws, size_t ws_size,
                              hipStream_t stream) {
    const float* xs   = (const float*)d_in[0];
    const float* xr   = (const float*)d_in[1];
    const int*   ei   = (const int*)  d_in[2];
    const float* dist = (const float*)d_in[3];
    const float* rot  = (const float*)d_in[4];
    const float* W1   = (const float*)d_in[5];
    const float* b1   = (const float*)d_in[6];
    const float* W2   = (const float*)d_in[7];
    const float* b2   = (const float*)d_in[8];

    int E = in_sizes[2] / 2;
    int N = in_sizes[0] / N_SCALARS;

    float* out = (float*)d_out;
    _Float16* W1t = (_Float16*)d_ws;                                   // 64 KB
    _Float16* W2t = (_Float16*)((char*)d_ws + MSG_IN * HIDDEN * 2);    // 24 KB

    zero_f32_kernel<<<(out_size + 255) / 256, 256, 0, stream>>>(out, out_size);
    convert_w_kernel<<<(MSG_IN * HIDDEN + 255) / 256, 256, 0, stream>>>(W1, W2, W1t, W2t);

    int nblocks = (E + TILE_E - 1) / TILE_E;
    eq_main_kernel<<<nblocks, 256, 0, stream>>>(xs, xr, ei, dist, rot,
                                                W1t, b1, W2t, b2, out, E, N);
}
